// EchoStateNetwork_28089086116126
// MI455X (gfx1250) — compile-verified, matrix operand order checked by reading, not run on hardware
//
#include <hip/hip_runtime.h>
#include <hip/hip_bf16.h>

// ---------------------------------------------------------------------------
// Echo State Network on MI455X (gfx1250, wave32, WMMA bf16)
//   B=32, S=4096, I=64, R=1024, O=8
// 16 persistent workgroups; each owns a 64-row W_res slice resident in LDS
// as bf16 (128 KB of the 320 KB WGP LDS). The 64 KB recurrent state
// ping-pongs through global scratch; one grid barrier per step.
// Software-pipelined WMMA K-loop, native v_tanh_f32, async global->LDS
// state staging (GLOBAL_LOAD_ASYNC_TO_LDS_B128 / ASYNCcnt), x prefetch.
// ---------------------------------------------------------------------------

#define ESN_B   32
#define ESN_S   4096
#define ESN_I   64
#define ESN_R   1024
#define ESN_O   8
#define ESN_F   (ESN_I + ESN_R)   // 1088
#define NWG     16
#define CHUNK   64                // R / NWG
#define NTHR    256               // 8 waves

typedef __attribute__((ext_vector_type(16))) __bf16 v16bf;
typedef __attribute__((ext_vector_type(8)))  float  v8f;
typedef __attribute__((__vector_size__(16))) int    v4i;
typedef __attribute__((address_space(1))) v4i*      gv4i_ptr;   // global
typedef __attribute__((address_space(3))) v4i*      lv4i_ptr;   // LDS

// ---------------------------------------------------------------------------
// LDS layout (dynamic), all offsets 1 KB aligned:
//   [0      .. 131072)  w_res_lds [64][1024] bf16   (this WG's W_res slice)
//   [131072 .. 196608)  r_old_lds [32][1024] bf16   (full old state, per step)
//   [196608 .. 204800)  w_in_lds  [64][64]   bf16   (this WG's W_in slice)
//   [204800 .. 208896)  x_lds     [32][64]   bf16   (x[:,t,:], per step)
//   [208896 .. 210944)  w_out_lds [8][64]    f32    (this WG's W_out slice)
//   [210944 .. 219136)  r_stage   [32][64]   f32    (new state chunk)
// ---------------------------------------------------------------------------
#define SMEM_BYTES 219136

__device__ __forceinline__ float esn_tanh(float v) {
#if __has_builtin(__builtin_amdgcn_tanhf)
    return __builtin_amdgcn_tanhf(v);      // native V_TANH_F32 (trans pipe)
#else
    return tanhf(v);
#endif
}

__device__ __forceinline__ void grid_barrier(unsigned* cnt, unsigned* gen) {
    __syncthreads();
    if (threadIdx.x == 0) {
        unsigned g = __hip_atomic_load(gen, __ATOMIC_ACQUIRE, __HIP_MEMORY_SCOPE_AGENT);
        unsigned a = __hip_atomic_fetch_add(cnt, 1u, __ATOMIC_ACQ_REL, __HIP_MEMORY_SCOPE_AGENT);
        if (a == NWG - 1u) {
            __hip_atomic_store(cnt, 0u, __ATOMIC_RELAXED, __HIP_MEMORY_SCOPE_AGENT);
            __hip_atomic_fetch_add(gen, 1u, __ATOMIC_RELEASE, __HIP_MEMORY_SCOPE_AGENT);
        } else {
            while (__hip_atomic_load(gen, __ATOMIC_ACQUIRE, __HIP_MEMORY_SCOPE_AGENT) == g) {
                __builtin_amdgcn_s_sleep(1);
            }
        }
    }
    __syncthreads();
}

// Zero barrier words and r0 (state buffer 0). rbuf holds 2*32*1024 bf16.
__global__ void esn_init_kernel(unsigned* cnt, unsigned* gen, unsigned short* rbuf_raw) {
    int g = blockIdx.x * blockDim.x + threadIdx.x;
    if (g == 0) { *cnt = 0u; *gen = 0u; }
    if (g < ESN_B * ESN_R) rbuf_raw[g] = 0;   // bf16 +0.0
}

// out[b,t,o] = sum_i x[b,t,i] * W_out[o,i]   (overwrites -> re-inits d_out)
__global__ void esn_out_x_kernel(const float* __restrict__ x,
                                 const float* __restrict__ W_out,
                                 float* __restrict__ out) {
    int g = blockIdx.x * blockDim.x + threadIdx.x;   // one (b,t) per thread
    if (g >= ESN_B * ESN_S) return;
    const float* xr = x + (size_t)g * ESN_I;
    float acc[ESN_O];
#pragma unroll
    for (int o = 0; o < ESN_O; ++o) acc[o] = 0.0f;
    for (int i = 0; i < ESN_I; ++i) {
        float xi = xr[i];
#pragma unroll
        for (int o = 0; o < ESN_O; ++o) acc[o] += xi * W_out[o * ESN_F + i];
    }
#pragma unroll
    for (int o = 0; o < ESN_O; ++o) out[(size_t)g * ESN_O + o] = acc[o];
}

// Persistent recurrence kernel: 16 WGs x 256 threads.
__global__ void esn_recurrence_kernel(const float* __restrict__ x,
                                      const float* __restrict__ W_in,
                                      const float* __restrict__ W_res,
                                      const float* __restrict__ W_out,
                                      float* __restrict__ out,
                                      unsigned short* __restrict__ rbuf_raw,
                                      unsigned* cnt, unsigned* gen) {
    extern __shared__ char smem[];
    __bf16 (*w_res_lds)[ESN_R] = (__bf16(*)[ESN_R])(smem);
    __bf16 (*r_old_lds)[ESN_R] = (__bf16(*)[ESN_R])(smem + 131072);
    __bf16 (*w_in_lds)[ESN_I]  = (__bf16(*)[ESN_I])(smem + 196608);
    __bf16 (*x_lds)[ESN_I]     = (__bf16(*)[ESN_I])(smem + 204800);
    float  (*w_out_lds)[CHUNK] = (float (*)[CHUNK])(smem + 208896);
    float  (*r_stage)[CHUNK]   = (float (*)[CHUNK])(smem + 210944);

    __bf16* rbuf = reinterpret_cast<__bf16*>(rbuf_raw);   // [2][32][1024]

    const int tid  = threadIdx.x;
    const int wg   = blockIdx.x;           // 0..15 -> rows wg*64 .. wg*64+63
    const int lane = tid & 31;
    const int wave = tid >> 5;             // 0..7
    const int m_base = (wave >> 2) * 16;   // batch-tile base: 0 or 16
    const int n_base = (wave & 3) * 16;    // chunk-column tile base: 0/16/32/48
    const int row_a  = lane & 15;
    const int khalf  = (lane >> 4) ? 16 : 0;

    // ---- one-time: load W_res / W_in slices (fp32 -> bf16) and W_out slice
    for (int idx = tid; idx < CHUNK * ESN_R; idx += NTHR) {
        int jl = idx >> 10, k = idx & (ESN_R - 1);
        w_res_lds[jl][k] = (__bf16)W_res[(size_t)(wg * CHUNK + jl) * ESN_R + k];
    }
    for (int idx = tid; idx < CHUNK * ESN_I; idx += NTHR) {
        int jl = idx >> 6, i = idx & (ESN_I - 1);
        w_in_lds[jl][i] = (__bf16)W_in[(size_t)(wg * CHUNK + jl) * ESN_I + i];
    }
    for (int idx = tid; idx < ESN_O * CHUNK; idx += NTHR) {
        int o = idx >> 6, jl = idx & (CHUNK - 1);
        w_out_lds[o][jl] = W_out[(size_t)o * ESN_F + ESN_I + wg * CHUNK + jl];
    }
    __syncthreads();

    for (int t = 0; t < ESN_S; ++t) {
        const __bf16* rbuf_old = rbuf + (size_t)(t & 1) * ESN_B * ESN_R;
        __bf16*       rbuf_new = rbuf + (size_t)((t + 1) & 1) * ESN_B * ESN_R;

        // ---- stage x[:,t,:] (fp32->bf16); prefetch next step's x slice
        for (int idx = tid; idx < ESN_B * ESN_I; idx += NTHR) {
            int b = idx >> 6, i = idx & (ESN_I - 1);
            const float* src = &x[((size_t)b * ESN_S + t) * ESN_I + i];
            x_lds[b][i] = (__bf16)(*src);
            if (t + 1 < ESN_S) __builtin_prefetch(src + ESN_I, 0, 3);
        }

        // ---- stage the full old state (64 KB) into LDS
#if __has_builtin(__builtin_amdgcn_global_load_async_to_lds_b128)
        {
            char* src = const_cast<char*>((const char*)rbuf_old);
            char* dst = (char*)&r_old_lds[0][0];
            for (int idx = tid; idx < (ESN_B * ESN_R * 2) / 16; idx += NTHR) {
                __builtin_amdgcn_global_load_async_to_lds_b128(
                    (gv4i_ptr)(src + idx * 16),
                    (lv4i_ptr)(dst + idx * 16),
                    0, 0);
            }
#if __has_builtin(__builtin_amdgcn_s_wait_asynccnt)
            __builtin_amdgcn_s_wait_asynccnt(0);
#else
            asm volatile("s_wait_asynccnt 0" ::: "memory");
#endif
        }
#else
        {
            const uint4* src = (const uint4*)rbuf_old;      // 64 KB, 16 B chunks
            uint4*       dst = (uint4*)&r_old_lds[0][0];
            for (int idx = tid; idx < (ESN_B * ESN_R * 2) / 16; idx += NTHR)
                dst[idx] = src[idx];
        }
#endif
        __syncthreads();

        // ---- fused pre-activation: acc = x_t @ W_in_sliceT + r_old @ W_res_sliceT
        v8f acc = {0.f, 0.f, 0.f, 0.f, 0.f, 0.f, 0.f, 0.f};
        {
            v16bf a0 = *(const v16bf*)&x_lds[m_base + row_a][khalf];
            v16bf b0 = *(const v16bf*)&w_in_lds[n_base + row_a][khalf];
            v16bf a1 = *(const v16bf*)&x_lds[m_base + row_a][32 + khalf];
            v16bf b1 = *(const v16bf*)&w_in_lds[n_base + row_a][32 + khalf];
            acc = __builtin_amdgcn_wmma_f32_16x16x32_bf16(
                      false, a0, false, b0, (short)0, acc, false, false);
            acc = __builtin_amdgcn_wmma_f32_16x16x32_bf16(
                      false, a1, false, b1, (short)0, acc, false, false);
        }
        {
            // software-pipelined K loop: fragments for chunk k loaded one
            // iteration ahead so ds_loads overlap the current WMMA
            const __bf16* arow = &r_old_lds[m_base + row_a][khalf];
            const __bf16* brow = &w_res_lds[n_base + row_a][khalf];
            v16bf a = *(const v16bf*)(arow);
            v16bf b = *(const v16bf*)(brow);
#pragma unroll 4
            for (int kk = 32; kk < ESN_R; kk += 32) {
                v16bf an = *(const v16bf*)(arow + kk);
                v16bf bn = *(const v16bf*)(brow + kk);
                acc = __builtin_amdgcn_wmma_f32_16x16x32_bf16(
                          false, a, false, b, (short)0, acc, false, false);
                a = an; b = bn;
            }
            acc = __builtin_amdgcn_wmma_f32_16x16x32_bf16(
                      false, a, false, b, (short)0, acc, false, false);
        }

        // ---- activation; stage new state chunk (C/D layout: VGPR e -> M=e(+8))
#pragma unroll
        for (int e = 0; e < 8; ++e) {
            float r = esn_tanh(acc[e]);
            int b  = m_base + e + ((lane >> 4) ? 8 : 0);
            int jl = n_base + (lane & 15);
            r_stage[b][jl] = r;
        }
        __syncthreads();

        // ---- publish bf16 state chunk to the ping-pong buffer
        for (int idx = tid; idx < ESN_B * CHUNK; idx += NTHR) {
            int b = idx >> 6, jl = idx & (CHUNK - 1);
            rbuf_new[(size_t)b * ESN_R + wg * CHUNK + jl] = (__bf16)r_stage[b][jl];
        }

        // ---- partial readout: one (b,o) pair per thread (32*8 == 256)
        {
            int b = tid >> 3, o = tid & 7;
            float s = 0.f;
#pragma unroll 8
            for (int jl = 0; jl < CHUNK; ++jl) s += r_stage[b][jl] * w_out_lds[o][jl];
            float* dst = &out[((size_t)b * ESN_S + t) * ESN_O + o];
            __hip_atomic_fetch_add(dst, s, __ATOMIC_RELAXED, __HIP_MEMORY_SCOPE_AGENT);
        }

        // ---- device-wide step barrier (release writes / acquire for next read)
        grid_barrier(cnt, gen);
    }
}

extern "C" void kernel_launch(void* const* d_in, const int* in_sizes, int n_in,
                              void* d_out, int out_size, void* d_ws, size_t ws_size,
                              hipStream_t stream) {
    const float* x     = (const float*)d_in[0];   // [32,4096,64]
    const float* W_in  = (const float*)d_in[1];   // [1024,64]
    const float* W_res = (const float*)d_in[2];   // [1024,1024]
    const float* W_out = (const float*)d_in[3];   // [8,1088]
    float* out = (float*)d_out;                   // [32,4096,8]

    unsigned char* ws = (unsigned char*)d_ws;
    unsigned* cnt = (unsigned*)(ws);
    unsigned* gen = (unsigned*)(ws + 128);
    unsigned short* rbuf = (unsigned short*)(ws + 256);   // 2*32*1024 bf16

    // zero barrier words + r0
    esn_init_kernel<<<(ESN_B * ESN_R + 255) / 256, 256, 0, stream>>>(cnt, gen, rbuf);
    // out <- x-part of readout (also re-initializes d_out deterministically)
    esn_out_x_kernel<<<(ESN_B * ESN_S + 255) / 256, 256, 0, stream>>>(x, W_out, out);
    // persistent recurrence: 16 WGs, 8 waves each, 219 KB LDS per WG
    esn_recurrence_kernel<<<NWG, NTHR, SMEM_BYTES, stream>>>(
        x, W_in, W_res, W_out, out, rbuf, cnt, gen);
}
